// Hydra_9363028706091
// MI455X (gfx1250) — matrix operand order repaired and checked by reading, not compiled, not run on hardware
//
#include <hip/hip_runtime.h>
#include <hip/hip_bf16.h>

typedef __attribute__((ext_vector_type(16))) _Float16 v16h;
typedef __attribute__((ext_vector_type(8)))  float    v8f;

#define LSEQ    4096
#define DMODEL  192
#define DINNER  384
#define NHEADSC 6
#define HEADD   64
#define DSTATE  64
#define CONVDIM 640
#define DINPROJ 1036
#define NCHUNK  16
#define CHUNKL  256
#define BATCH   4
#define BATCH2  8

// ---------------------------------------------------------------------------
// WMMA helpers: v_wmma_f32_16x16x32_f16, D = A(16x32) * B(32x16) + C
// A layout: lane&15 = M row; elems 0..7 -> K = kb..kb+7, elems 8..15 -> K = kb+8..kb+15
//           with kb = k0 + (lane>>4)*16
// B layout: lane&15 = N col; elem e -> K = k0 + (lane>>4)*16 + e
// C/D:      lane&15 = N col; vgpr r -> M = r + 8*(lane>>4)
// ---------------------------------------------------------------------------
__device__ __forceinline__ v8f wmma_f16(v16h a, v16h b, v8f c) {
  return __builtin_amdgcn_wmma_f32_16x16x32_f16(false, a, false, b, (short)0, c,
                                                false, false);
}

__device__ __forceinline__ v16h fragA_contig(const float* __restrict__ rp, int k0,
                                             int lane) {
  int kb = k0 + ((lane >> 4) << 4);
  v16h f;
#pragma unroll
  for (int i = 0; i < 8; ++i) {
    f[i]     = (_Float16)rp[kb + i];
    f[i + 8] = (_Float16)rp[kb + 8 + i];
  }
  return f;
}

__device__ __forceinline__ v16h fragB_contig(const float* __restrict__ cp, int k0,
                                             int lane) {
  int kb = k0 + ((lane >> 4) << 4);
  v16h f;
#pragma unroll
  for (int e = 0; e < 16; ++e) f[e] = (_Float16)cp[kb + e];
  return f;
}

// Row accessor into conv output xbc (4, 4096, 640) with direction flip for bp>=4
__device__ __forceinline__ const float* xbc_row(const float* __restrict__ xbc,
                                                int bp, int l) {
  int b  = bp & 3;
  int ll = (bp < BATCH) ? l : (LSEQ - 1 - l);
  return xbc + ((size_t)(b * LSEQ + ll)) * CONVDIM;
}

// ---------------------------------------------------------------------------
// K0: transpose u (4,192,4096) -> x_in (4,4096,192)
// ---------------------------------------------------------------------------
__global__ void k_transpose_u(const float* __restrict__ u, float* __restrict__ xin) {
  int idx = blockIdx.x * blockDim.x + threadIdx.x;
  if (idx >= BATCH * LSEQ * DMODEL) return;
  int d = idx % DMODEL;
  int l = (idx / DMODEL) % LSEQ;
  int b = idx / (DMODEL * LSEQ);
  xin[idx] = u[((size_t)(b * DMODEL + d)) * LSEQ + l];
}

// ---------------------------------------------------------------------------
// Generic WMMA GEMM: C[M,N] = A[M,K] * W[N,K]^T  (both f32 in memory, f16 frags)
// TRANS=1: store C[(b*N+n)*ldc_l + l] with b=m/ldc_l, l=m%ldc_l (out_proj)
// ---------------------------------------------------------------------------
template <int TRANS>
__global__ void k_gemm(const float* __restrict__ A, const float* __restrict__ W,
                       float* __restrict__ C, int M, int N, int K, int ldc_l) {
  int lane   = threadIdx.x & 31;
  int wave   = threadIdx.x >> 5;
  int ntiles = (N + 15) >> 4;
  int mtiles = M >> 4;
  long tile  = (long)blockIdx.x * 4 + wave;
  if (tile >= (long)ntiles * mtiles) return;
  int tm = (int)(tile / ntiles), tn = (int)(tile % ntiles);
  int m  = (tm << 4) + (lane & 15);
  int n  = (tn << 4) + (lane & 15);
  int nc = (n < N) ? n : (N - 1);
  const float* arow = A + (size_t)m * K;
  const float* brow = W + (size_t)nc * K;
  v8f acc = {};
  for (int k0 = 0; k0 < K; k0 += 32) {
    __builtin_prefetch(arow + k0 + 64, 0, 1);
    v16h fa = fragA_contig(arow, k0, lane);
    v16h fb = fragB_contig(brow, k0, lane);
    acc = wmma_f16(fa, fb, acc);
  }
#pragma unroll
  for (int r = 0; r < 8; ++r) {
    int mm = (tm << 4) + r + ((lane >> 4) << 3);
    if (n < N) {
      if (TRANS) {
        int b = mm / ldc_l, l = mm % ldc_l;
        C[((size_t)(b * N + n)) * ldc_l + l] = acc[r];
      } else {
        C[(size_t)mm * N + n] = acc[r];
      }
    }
  }
}

// ---------------------------------------------------------------------------
// K2: dt = softplus(dt_raw(+flip for bw) + dt_bias)  -> (8,4096,6)
// ---------------------------------------------------------------------------
__global__ void k_dt(const float* __restrict__ zx, const float* __restrict__ dt_bias,
                     float* __restrict__ dt) {
  int idx = blockIdx.x * blockDim.x + threadIdx.x;
  if (idx >= BATCH2 * LSEQ * NHEADSC) return;
  int h  = idx % NHEADSC;
  int l  = (idx / NHEADSC) % LSEQ;
  int bp = idx / (NHEADSC * LSEQ);
  float v;
  if (bp < BATCH)
    v = zx[((size_t)(bp * LSEQ + l)) * DINPROJ + (DINNER + CONVDIM) + h];
  else
    v = zx[((size_t)((bp - BATCH) * LSEQ + (LSEQ - 1 - l))) * DINPROJ +
           (DINNER + CONVDIM) + NHEADSC + h];
  v += dt_bias[h];
  dt[idx] = (v > 20.f) ? v : log1pf(expf(v));
}

// ---------------------------------------------------------------------------
// K2b: per-chunk cumsum of dt*A  -> cum[((bp*16+c)*6+h)*256 + t]
// ---------------------------------------------------------------------------
__global__ void k_cumsum(const float* __restrict__ dt, const float* __restrict__ A_log,
                         float* __restrict__ cum) {
  int idx = blockIdx.x * blockDim.x + threadIdx.x;
  if (idx >= BATCH2 * NCHUNK * NHEADSC) return;
  int h  = idx % NHEADSC;
  int c  = (idx / NHEADSC) % NCHUNK;
  int bp = idx / (NHEADSC * NCHUNK);
  float Aneg = -expf(A_log[h]);
  float acc  = 0.f;
  size_t base = (size_t)idx * CHUNKL;
  for (int t = 0; t < CHUNKL; ++t) {
    acc += dt[((size_t)(bp * LSEQ) + c * CHUNKL + t) * NHEADSC + h] * Aneg;
    cum[base + t] = acc;
  }
}

// ---------------------------------------------------------------------------
// K3: depthwise conv7 (zero pad 3) + bias + SiLU over xBC slice of zxbcdt
// ---------------------------------------------------------------------------
__global__ void k_conv(const float* __restrict__ zx, const float* __restrict__ cw,
                       const float* __restrict__ cb, float* __restrict__ xbc) {
  int idx = blockIdx.x * blockDim.x + threadIdx.x;
  if (idx >= BATCH * LSEQ * CONVDIM) return;
  int ch = idx % CONVDIM;
  int l  = (idx / CONVDIM) % LSEQ;
  int b  = idx / (CONVDIM * LSEQ);
  float acc = cb[ch];
#pragma unroll
  for (int k = 0; k < 7; ++k) {
    int ll = l + k - 3;
    if (ll >= 0 && ll < LSEQ)
      acc += zx[((size_t)(b * LSEQ + ll)) * DINPROJ + DINNER + ch] * cw[ch * 7 + k];
  }
  xbc[idx] = acc / (1.f + expf(-acc));
}

// ---------------------------------------------------------------------------
// K4: CB[t,s] = sum_n C[t,n]*B[s,n] per (bp,chunk), lower-triangular tiles only
// ---------------------------------------------------------------------------
__global__ void k_cb(const float* __restrict__ xbc, float* __restrict__ cb_ws) {
  int bp   = blockIdx.x / NCHUNK;
  int c    = blockIdx.x % NCHUNK;
  int lane = threadIdx.x & 31;
  int wave = threadIdx.x >> 5;
  int boff = (bp < BATCH) ? DINNER : (DINNER + 2 * DSTATE);
  int coff = boff + DSTATE;
  float* cbout = cb_ws + (size_t)blockIdx.x * (CHUNKL * CHUNKL);
  for (int tile = wave; tile < 256; tile += 8) {
    int tt = tile >> 4, ts = tile & 15;
    if (tt < ts) continue;  // uniform per wave
    int t = (tt << 4) + (lane & 15);
    int s = (ts << 4) + (lane & 15);
    const float* arow = xbc_row(xbc, bp, c * CHUNKL + t) + coff;  // Cr
    const float* brow = xbc_row(xbc, bp, c * CHUNKL + s) + boff;  // Br
    v8f acc = {};
#pragma unroll
    for (int k0 = 0; k0 < DSTATE; k0 += 32) {
      v16h fa = fragA_contig(arow, k0, lane);
      v16h fb = fragB_contig(brow, k0, lane);
      acc = wmma_f16(fa, fb, acc);
    }
#pragma unroll
    for (int r = 0; r < 8; ++r) {
      int ttt = (tt << 4) + r + ((lane >> 4) << 3);
      cbout[ttt * CHUNKL + s] = acc[r];
    }
  }
}

// ---------------------------------------------------------------------------
// K5: states[h,p,n] = sum_s B[s,n] * dt[s]*exp(cum_last-cum[s]) * x[s,p]
// ---------------------------------------------------------------------------
__global__ void k_states(const float* __restrict__ xbc, const float* __restrict__ dt,
                         const float* __restrict__ cum, float* __restrict__ states) {
  int h    = blockIdx.x % NHEADSC;
  int c    = (blockIdx.x / NHEADSC) % NCHUNK;
  int bp   = blockIdx.x / (NHEADSC * NCHUNK);
  int lane = threadIdx.x & 31, wave = threadIdx.x >> 5;
  int boff = (bp < BATCH) ? DINNER : (DINNER + 2 * DSTATE);
  const float* cumr = cum + (size_t)blockIdx.x * CHUNKL;
  float cl = cumr[CHUNKL - 1];
  float* sout = states + (size_t)blockIdx.x * (HEADD * DSTATE);
  for (int tile = wave; tile < 16; tile += 4) {
    int tp = tile >> 2, tn = tile & 3;
    int p  = (tp << 4) + (lane & 15);
    int n  = (tn << 4) + (lane & 15);
    v8f acc = {};
    for (int k0 = 0; k0 < CHUNKL; k0 += 32) {
      int kb = k0 + ((lane >> 4) << 4);
      v16h fa;
#pragma unroll
      for (int i = 0; i < 8; ++i) {
        int s1 = kb + i, s2 = kb + 8 + i;
        float w1 = dt[((size_t)(bp * LSEQ) + c * CHUNKL + s1) * NHEADSC + h] *
                   expf(cl - cumr[s1]);
        float w2 = dt[((size_t)(bp * LSEQ) + c * CHUNKL + s2) * NHEADSC + h] *
                   expf(cl - cumr[s2]);
        fa[i]     = (_Float16)(xbc_row(xbc, bp, c * CHUNKL + s1)[h * HEADD + p] * w1);
        fa[i + 8] = (_Float16)(xbc_row(xbc, bp, c * CHUNKL + s2)[h * HEADD + p] * w2);
      }
      int kb2 = k0 + ((lane >> 4) << 4);
      v16h fb;
#pragma unroll
      for (int e = 0; e < 16; ++e)
        fb[e] = (_Float16)(xbc_row(xbc, bp, c * CHUNKL + kb2 + e)[boff + n]);
      acc = wmma_f16(fa, fb, acc);
    }
#pragma unroll
    for (int r = 0; r < 8; ++r) {
      int pp = (tp << 4) + r + ((lane >> 4) << 3);
      sout[pp * DSTATE + n] = acc[r];
    }
  }
}

// ---------------------------------------------------------------------------
// K6: exclusive inter-chunk recurrence: prev[c] = state before chunk c
// ---------------------------------------------------------------------------
__global__ void k_recur(const float* __restrict__ states, const float* __restrict__ cum,
                        float* __restrict__ prevst) {
  int h   = blockIdx.x % NHEADSC;
  int bp  = blockIdx.x / NHEADSC;
  int tid = threadIdx.x;  // 256 threads, 16 elems each
  float prev[16];
#pragma unroll
  for (int j = 0; j < 16; ++j) prev[j] = 0.f;
  for (int c = 0; c < NCHUNK; ++c) {
    size_t chunkIdx = (size_t)((bp * NCHUNK + c) * NHEADSC + h);
    size_t base     = chunkIdx * (HEADD * DSTATE);
    float  cd       = expf(cum[chunkIdx * CHUNKL + CHUNKL - 1]);
#pragma unroll
    for (int j = 0; j < 16; ++j) {
      size_t e = base + tid + 256 * j;
      prevst[e] = prev[j];
      prev[j]   = cd * prev[j] + states[e];
    }
  }
}

// ---------------------------------------------------------------------------
// K7: y = Yd + Yo per (bp,chunk,head)
//  Yd[t,p] = sum_{s<=t} CB[t,s]*exp(cum_t-cum_s)*dt[s] * x[s,p]   (WMMA)
//  Yo[t,p] = exp(cum_t) * sum_n C[t,n]*prev[p,n]                  (WMMA)
// ---------------------------------------------------------------------------
__global__ void k_ydyo(const float* __restrict__ xbc, const float* __restrict__ dt,
                       const float* __restrict__ cum, const float* __restrict__ cb_ws,
                       const float* __restrict__ prevst, float* __restrict__ y) {
  int h    = blockIdx.x % NHEADSC;
  int c    = (blockIdx.x / NHEADSC) % NCHUNK;
  int bp   = blockIdx.x / (NHEADSC * NCHUNK);
  int lane = threadIdx.x & 31, wave = threadIdx.x >> 5;
  int boff = (bp < BATCH) ? DINNER : (DINNER + 2 * DSTATE);
  int coff = boff + DSTATE;
  const float* cumr = cum + (size_t)blockIdx.x * CHUNKL;
  const float* cbin = cb_ws + (size_t)(bp * NCHUNK + c) * (CHUNKL * CHUNKL);
  const float* pst  = prevst + (size_t)blockIdx.x * (HEADD * DSTATE);
  v8f zero = {};
  for (int tt = wave; tt < 16; tt += 8) {
    int   t    = (tt << 4) + (lane & 15);
    float cumt = cumr[t];
    v8f accd[4] = {zero, zero, zero, zero};
    for (int ks = 0; ks <= (tt >> 1); ++ks) {  // lower-triangular K sweep
      int k0 = ks * 32;
      int kb = k0 + ((lane >> 4) << 4);
      v16h fa;
#pragma unroll
      for (int i = 0; i < 8; ++i) {
        int   s1 = kb + i, s2 = kb + 8 + i;
        float v1 = 0.f, v2 = 0.f;
        if (s1 <= t)
          v1 = cbin[t * CHUNKL + s1] * expf(cumt - cumr[s1]) *
               dt[((size_t)(bp * LSEQ) + c * CHUNKL + s1) * NHEADSC + h];
        if (s2 <= t)
          v2 = cbin[t * CHUNKL + s2] * expf(cumt - cumr[s2]) *
               dt[((size_t)(bp * LSEQ) + c * CHUNKL + s2) * NHEADSC + h];
        fa[i]     = (_Float16)v1;
        fa[i + 8] = (_Float16)v2;
      }
#pragma unroll
      for (int pt = 0; pt < 4; ++pt) {
        int  p   = (pt << 4) + (lane & 15);
        int  kb2 = k0 + ((lane >> 4) << 4);
        v16h fb;
#pragma unroll
        for (int e = 0; e < 16; ++e)
          fb[e] = (_Float16)(xbc_row(xbc, bp, c * CHUNKL + kb2 + e)[h * HEADD + p]);
        accd[pt] = wmma_f16(fa, fb, accd[pt]);
      }
    }
    // Yo: A = Cr (K=n), B = prev[p,n]
    v8f acco[4] = {zero, zero, zero, zero};
    const float* arow = xbc_row(xbc, bp, c * CHUNKL + t) + coff;
#pragma unroll
    for (int k0 = 0; k0 < DSTATE; k0 += 32) {
      v16h fa = fragA_contig(arow, k0, lane);
#pragma unroll
      for (int pt = 0; pt < 4; ++pt) {
        int  p   = (pt << 4) + (lane & 15);
        int  kb2 = k0 + ((lane >> 4) << 4);
        v16h fb;
#pragma unroll
        for (int e = 0; e < 16; ++e) fb[e] = (_Float16)pst[p * DSTATE + kb2 + e];
        acco[pt] = wmma_f16(fa, fb, acco[pt]);
      }
    }
#pragma unroll
    for (int r = 0; r < 8; ++r) {
      int   ttt = (tt << 4) + r + ((lane >> 4) << 3);
      float et  = expf(cumr[ttt]);
      int   l   = c * CHUNKL + ttt;
#pragma unroll
      for (int pt = 0; pt < 4; ++pt) {
        int p = (pt << 4) + (lane & 15);
        y[((size_t)(bp * LSEQ + l)) * DINNER + h * HEADD + p] =
            accd[pt][r] + et * acco[pt][r];
      }
    }
  }
}

// ---------------------------------------------------------------------------
// K8: roll/flip combine + dskip + RMS-norm + SiLU(z) gate -> y_final (4,4096,384)
// ---------------------------------------------------------------------------
__global__ void k_combine(const float* __restrict__ y, const float* __restrict__ xbc,
                          const float* __restrict__ zx, const float* __restrict__ fcD,
                          const float* __restrict__ Dp, const float* __restrict__ normw,
                          float* __restrict__ yfin) {
  __shared__ float sdata[DINNER];
  __shared__ float sds[NHEADSC];
  int bl = blockIdx.x;  // b*4096 + l
  int b = bl / LSEQ, l = bl % LSEQ;
  int d = threadIdx.x;  // 384
  float xog = xbc[((size_t)bl) * CONVDIM + d];
  // dskip[h] = dot(fc_D_w[h], x_og) + D[h]
  for (int h = 0; h < NHEADSC; ++h) {
    sdata[d] = xog * fcD[h * DINNER + d];
    __syncthreads();
    if (d < 128) sdata[d] += sdata[d + 128] + sdata[d + 256];
    __syncthreads();
    for (int st = 64; st > 0; st >>= 1) {
      if (d < st) sdata[d] += sdata[d + st];
      __syncthreads();
    }
    if (d == 0) sds[h] = sdata[0] + Dp[h];
    __syncthreads();
  }
  float yf = (l == 0) ? 0.f : y[((size_t)(b * LSEQ + l - 1)) * DINNER + d];
  float yb = (l == LSEQ - 1)
                 ? 0.f
                 : y[((size_t)((b + BATCH) * LSEQ + (LSEQ - 2 - l))) * DINNER + d];
  float yt = yf + yb + xog * sds[d >> 6];
  sdata[d] = yt * yt;
  __syncthreads();
  if (d < 128) sdata[d] += sdata[d + 128] + sdata[d + 256];
  __syncthreads();
  for (int st = 64; st > 0; st >>= 1) {
    if (d < st) sdata[d] += sdata[d + st];
    __syncthreads();
  }
  float ms = sdata[0] / (float)DINNER;
  float yn = yt * rsqrtf(ms + 1e-5f) * normw[d];
  float z  = zx[((size_t)bl) * DINPROJ + d];
  yfin[((size_t)bl) * DINNER + d] = yn * (z / (1.f + expf(-z)));
}

// ---------------------------------------------------------------------------
extern "C" void kernel_launch(void* const* d_in, const int* in_sizes, int n_in,
                              void* d_out, int out_size, void* d_ws, size_t ws_size,
                              hipStream_t stream) {
  const float* u       = (const float*)d_in[0];
  const float* W_in    = (const float*)d_in[1];
  const float* conv_w  = (const float*)d_in[2];
  const float* conv_b  = (const float*)d_in[3];
  const float* dt_bias = (const float*)d_in[4];
  const float* A_log   = (const float*)d_in[5];
  const float* Dp      = (const float*)d_in[6];
  const float* fc_D_w  = (const float*)d_in[7];
  const float* norm_w  = (const float*)d_in[8];
  const float* W_out   = (const float*)d_in[9];
  float* out = (float*)d_out;

  float* ws   = (float*)d_ws;
  float* xin  = ws;                    // 4*4096*192   = 3,145,728
  float* zx   = xin + 3145728;         // 4*4096*1036  = 16,973,824
  float* xbc  = zx + 16973824;         // 4*4096*640   = 10,485,760
  float* dtv  = xbc + 10485760;        // 8*4096*6     = 196,608
  float* cumv = dtv + 196608;          // 8*16*6*256   = 196,608
  float* cbws = cumv + 196608;         // 8*16*256*256 = 8,388,608
  float* stw  = cbws + 8388608;        // 8*16*6*64*64 = 3,145,728
  float* pstw = stw + 3145728;         // 3,145,728
  float* ybuf = pstw + 3145728;        // 8*4096*384   = 12,582,912
  float* yfin = ybuf + 12582912;       // 4*4096*384   = 6,291,456

  // K0: transpose u -> x_in
  k_transpose_u<<<(BATCH * LSEQ * DMODEL + 255) / 256, 256, 0, stream>>>(u, xin);

  // K1: in_proj GEMM (16384 x 1036 x 192)
  {
    int M = BATCH * LSEQ, N = DINPROJ, K = DMODEL;
    long tiles = (long)(M / 16) * ((N + 15) / 16);
    k_gemm<0><<<(int)((tiles + 3) / 4), 128, 0, stream>>>(xin, W_in, zx, M, N, K, 1);
  }

  // K2: dt softplus (with direction flip)
  k_dt<<<(BATCH2 * LSEQ * NHEADSC + 255) / 256, 256, 0, stream>>>(zx, dt_bias, dtv);

  // K2b: per-chunk cumsum of dt*A
  k_cumsum<<<(BATCH2 * NCHUNK * NHEADSC + 255) / 256, 256, 0, stream>>>(dtv, A_log,
                                                                        cumv);

  // K3: depthwise conv + SiLU
  k_conv<<<(BATCH * LSEQ * CONVDIM + 255) / 256, 256, 0, stream>>>(zx, conv_w, conv_b,
                                                                   xbc);

  // K4: CB per (bp, chunk)
  k_cb<<<BATCH2 * NCHUNK, 256, 0, stream>>>(xbc, cbws);

  // K5: per-chunk states
  k_states<<<BATCH2 * NCHUNK * NHEADSC, 128, 0, stream>>>(xbc, dtv, cumv, stw);

  // K6: inter-chunk recurrence (exclusive prefix of decayed states)
  k_recur<<<BATCH2 * NHEADSC, 256, 0, stream>>>(stw, cumv, pstw);

  // K7: Yd + Yo -> y
  k_ydyo<<<BATCH2 * NCHUNK * NHEADSC, 256, 0, stream>>>(xbc, dtv, cumv, cbws, pstw,
                                                        ybuf);

  // K8: combine, dskip, RMS-norm, gate
  k_combine<<<BATCH * LSEQ, DINNER, 0, stream>>>(ybuf, xbc, zx, fc_D_w, Dp, norm_w,
                                                 yfin);

  // K9: out_proj GEMM with transposed (b, e, l) store
  {
    int M = BATCH * LSEQ, N = DMODEL, K = DINNER;
    long tiles = (long)(M / 16) * ((N + 15) / 16);
    k_gemm<1><<<(int)((tiles + 3) / 4), 128, 0, stream>>>(yfin, W_out, out, M, N, K,
                                                          LSEQ);
  }
}
